// FeaturePropagationNeuralOperatorSeq_2989297238653
// MI455X (gfx1250) — matrix operand order repaired and checked by reading, not compiled, not run on hardware
//
#include <hip/hip_runtime.h>
#include <hip/hip_bf16.h>
#include <math.h>

// ---------------------------------------------------------------------------
// Problem constants (from the reference)
// ---------------------------------------------------------------------------
#define BATCH 4
#define DIMS  3
#define MPAR  64
#define N2 512
#define N1 2048
#define N0 8192
#define C2 256
#define C1 128
#define C0 64

typedef __attribute__((ext_vector_type(16))) _Float16 f16x16;
typedef __attribute__((ext_vector_type(8)))  _Float16 f16x8;
typedef __attribute__((ext_vector_type(8)))  float    f32x8;

// ---------------------------------------------------------------------------
// Per-sample parameter vectors: p = tanh(par @ Wp + bp)
//   par: (B, M)   Wp0: (M,128) Wp1: (M,64)
// ---------------------------------------------------------------------------
__global__ void par_vec_kernel(const float* __restrict__ par,
                               const float* __restrict__ Wp0, const float* __restrict__ bp0,
                               const float* __restrict__ Wp1, const float* __restrict__ bp1,
                               float* __restrict__ p0, float* __restrict__ p1)
{
    int o = blockIdx.x * blockDim.x + threadIdx.x;   // 4 * (128+64) = 768 outputs
    if (o >= BATCH * (128 + 64)) return;
    int b = o / 192;
    int j = o % 192;
    const float* pb = par + b * MPAR;
    float acc;
    if (j < 128) {
        acc = bp0[j];
        #pragma unroll 8
        for (int m = 0; m < MPAR; ++m) acc += pb[m] * Wp0[m * 128 + j];
        p0[b * 128 + j] = tanhf(acc);
    } else {
        int j2 = j - 128;
        acc = bp1[j2];
        #pragma unroll 8
        for (int m = 0; m < MPAR; ++m) acc += pb[m] * Wp1[m * 64 + j2];
        p1[b * 64 + j2] = tanhf(acc);
    }
}

// ---------------------------------------------------------------------------
// kNN (k=3) per target thread: scan all sources in the same sample, keep the
// three smallest d2, emit global gather indices + inverse-d2 weights.
// ---------------------------------------------------------------------------
__global__ void knn3_kernel(const float* __restrict__ posSrc,
                            const float* __restrict__ posTgt,
                            int nSrcPerSample, int nTgtPerSample, int nTgtTotal,
                            int* __restrict__ idxOut, float* __restrict__ wOut)
{
    int t = blockIdx.x * blockDim.x + threadIdx.x;
    if (t >= nTgtTotal) return;
    int b = t / nTgtPerSample;
    const float* ps = posSrc + (size_t)b * nSrcPerSample * DIMS;
    float tx = posTgt[t * 3 + 0];
    float ty = posTgt[t * 3 + 1];
    float tz = posTgt[t * 3 + 2];
    float d0 = 3.4e38f, d1 = 3.4e38f, d2 = 3.4e38f;
    int   i0 = 0, i1 = 0, i2 = 0;
    for (int s = 0; s < nSrcPerSample; ++s) {
        float dx = ps[s * 3 + 0] - tx;
        float dy = ps[s * 3 + 1] - ty;
        float dz = ps[s * 3 + 2] - tz;
        float d = dx * dx + dy * dy + dz * dz;
        if (d < d0)      { d2 = d1; i2 = i1; d1 = d0; i1 = i0; d0 = d; i0 = s; }
        else if (d < d1) { d2 = d1; i2 = i1; d1 = d;  i1 = s; }
        else if (d < d2) { d2 = d;  i2 = s; }
    }
    float w0 = 1.0f / fmaxf(d0, 1e-16f);
    float w1 = 1.0f / fmaxf(d1, 1e-16f);
    float w2 = 1.0f / fmaxf(d2, 1e-16f);
    int base = b * nSrcPerSample;
    idxOut[t * 3 + 0] = base + i0;
    idxOut[t * 3 + 1] = base + i1;
    idxOut[t * 3 + 2] = base + i2;
    wOut[t * 4 + 0] = w0;
    wOut[t * 4 + 1] = w1;
    wOut[t * 4 + 2] = w2;
    wOut[t * 4 + 3] = w0 + w1 + w2;
}

// ---------------------------------------------------------------------------
// Interpolate + concat skip features, writing the GEMM A operand in f16
// row-major [nTgt x (Cin+Cskip)].
// ---------------------------------------------------------------------------
__global__ void interp_concat_kernel(const float* __restrict__ feat, int Cin,
                                     const int* __restrict__ idx,
                                     const float* __restrict__ w,
                                     const float* __restrict__ skip, int Cskip,
                                     _Float16* __restrict__ out, int nTgt)
{
    int Ctot = Cin + Cskip;
    long total = (long)nTgt * Ctot;
    long e = (long)blockIdx.x * blockDim.x + threadIdx.x;
    if (e >= total) return;
    int row = (int)(e / Ctot);
    int col = (int)(e % Ctot);
    float v;
    if (col < Cin) {
        const int*   ip = idx + row * 3;
        const float* wp = w + row * 4;
        v = (wp[0] * feat[(size_t)ip[0] * Cin + col] +
             wp[1] * feat[(size_t)ip[1] * Cin + col] +
             wp[2] * feat[(size_t)ip[2] * Cin + col]) / wp[3];
    } else {
        v = skip[(size_t)row * Cskip + (col - Cin)];
    }
    out[e] = (_Float16)v;
}

// ---------------------------------------------------------------------------
// Pack a (Kd x Nd) f32 row-major weight matrix into the WMMA B-operand layout:
// per (ktile, ntile, lane) a contiguous 16-half chunk, where lane = hi*16+n,
// halves i -> B[ktile*32 + 16*hi + i][ntile*16 + n].
// ---------------------------------------------------------------------------
__global__ void pack_b_kernel(const float* __restrict__ W, _Float16* __restrict__ out,
                              int Kd, int Nd)
{
    int t = blockIdx.x * blockDim.x + threadIdx.x;
    if (t >= Kd * Nd) return;
    int i    = t & 15;
    int lane = (t >> 4) & 31;
    int rest = t >> 9;
    int tilesN = Nd >> 4;
    int nt = rest % tilesN;
    int kt = rest / tilesN;
    int hi = lane >> 4;
    int n  = nt * 16 + (lane & 15);
    int k  = kt * 32 + 16 * hi + i;
    out[t] = (_Float16)W[(size_t)k * Nd + n];
}

// ---------------------------------------------------------------------------
// WMMA GEMM, N-blocked: one wave computes a 16x64 output strip with four
// independent f32 accumulators sharing a single A operand per K-step.
//   Per K-step: 2 aligned b128 A loads + 4 b256 (2x b128) B loads feed
//   4 independent v_wmma_f32_16x16x32_f16 -> accumulation chains interleave,
//   hiding the WMMA D->C latency without v_nop padding, and A traffic /4.
//   A-operand per lane: halves i -> A[row][k0 + 16*hi + i]
//   C/D: c[r] <-> (m = r + 8*hi, n = lane&15)
// Epilogue: + bias[col]; optional tanh; optional per-sample scale par[s*N+col];
// write f16 (next GEMM input) and/or f32.
// ---------------------------------------------------------------------------
__global__ void wmma_gemm_kernel(const _Float16* __restrict__ A,
                                 const _Float16* __restrict__ Bp,
                                 const float* __restrict__ bias,
                                 const float* __restrict__ par, int rowsPerSample,
                                 _Float16* __restrict__ outH,
                                 float* __restrict__ outF,
                                 int M, int N, int Kd, int applyTanh)
{
    int wavesPerBlock = blockDim.x >> 5;
    int waveId = blockIdx.x * wavesPerBlock + (threadIdx.x >> 5);
    int lane = threadIdx.x & 31;
    int tilesN4 = N >> 6;                 // 64-wide strips
    int tileM  = waveId / tilesN4;
    int tileN4 = waveId % tilesN4;
    if (tileM * 16 >= M) return;          // wave-uniform: EXEC stays all-1s

    int hi = lane >> 4;
    int ln = lane & 15;
    const _Float16* Arow = A + (size_t)(tileM * 16 + ln) * Kd;
    int tilesN = N >> 4;
    const _Float16* Bt = Bp + ((size_t)tileN4 * 4) * 512 + (size_t)lane * 16;
    size_t bK = (size_t)tilesN * 512;     // halves per K-tile row of packed B

    f32x8 c0 = {}, c1 = {}, c2 = {}, c3 = {};
    int ktiles = Kd >> 5;
    for (int kt = 0; kt < ktiles; ++kt) {
        int k0 = kt * 32;
        f16x8 alo = *(const f16x8*)(Arow + k0 + 16 * hi);
        f16x8 ahi = *(const f16x8*)(Arow + k0 + 16 * hi + 8);
        f16x16 a;
        #pragma unroll
        for (int i = 0; i < 8; ++i) { a[i] = alo[i]; a[i + 8] = ahi[i]; }
        const _Float16* bbase = Bt + (size_t)kt * bK;
        f16x16 b0 = *(const f16x16*)(bbase);
        f16x16 b1 = *(const f16x16*)(bbase + 512);
        f16x16 b2 = *(const f16x16*)(bbase + 1024);
        f16x16 b3 = *(const f16x16*)(bbase + 1536);
        if (kt + 1 < ktiles) __builtin_prefetch(Arow + k0 + 32 + 8 * hi, 0, 1);
        c0 = __builtin_amdgcn_wmma_f32_16x16x32_f16(false, a, false, b0, (short)0, c0, false, false);
        c1 = __builtin_amdgcn_wmma_f32_16x16x32_f16(false, a, false, b1, (short)0, c1, false, false);
        c2 = __builtin_amdgcn_wmma_f32_16x16x32_f16(false, a, false, b2, (short)0, c2, false, false);
        c3 = __builtin_amdgcn_wmma_f32_16x16x32_f16(false, a, false, b3, (short)0, c3, false, false);
    }

    f32x8 cc[4] = { c0, c1, c2, c3 };
    #pragma unroll
    for (int ns = 0; ns < 4; ++ns) {
        int col = tileN4 * 64 + ns * 16 + ln;
        float bv = bias ? bias[col] : 0.0f;
        #pragma unroll
        for (int r = 0; r < 8; ++r) {
            int m = tileM * 16 + r + 8 * hi;
            float v = cc[ns][r] + bv;
            if (applyTanh) v = tanhf(v);
            if (par) {
                int s = m / rowsPerSample;
                v *= par[s * N + col];
            }
            if (outH) outH[(size_t)m * N + col] = (_Float16)v;
            if (outF) outF[(size_t)m * N + col] = v;
        }
    }
}

// ---------------------------------------------------------------------------
// Host-side launch
// ---------------------------------------------------------------------------
static inline void* bump(uintptr_t& cur, size_t bytes)
{
    cur = (cur + 255) & ~(uintptr_t)255;
    void* p = (void*)cur;
    cur += bytes;
    return p;
}

extern "C" void kernel_launch(void* const* d_in, const int* in_sizes, int n_in,
                              void* d_out, int out_size, void* d_ws, size_t ws_size,
                              hipStream_t stream)
{
    const float* par        = (const float*)d_in[0];
    const float* x          = (const float*)d_in[1];   // (B*N2, C2)
    const float* pos        = (const float*)d_in[2];   // (B*N2, 3)
    const float* x_skip_l1  = (const float*)d_in[4];   // (B*N1, C1)
    const float* pos_sk_l1  = (const float*)d_in[5];   // (B*N1, 3)
    const float* x_skip_l0  = (const float*)d_in[7];   // (B*N0, C0)
    const float* pos_sk_l0  = (const float*)d_in[8];   // (B*N0, 3)
    const void*  batch_l0   = d_in[9];                 // (B*N0,) int32
    const float* W0a = (const float*)d_in[10]; const float* b0a = (const float*)d_in[11];
    const float* W0b = (const float*)d_in[12]; const float* b0b = (const float*)d_in[13];
    const float* Wp0 = (const float*)d_in[14]; const float* bp0 = (const float*)d_in[15];
    const float* W1a = (const float*)d_in[16]; const float* b1a = (const float*)d_in[17];
    const float* W1b = (const float*)d_in[18]; const float* b1b = (const float*)d_in[19];
    const float* Wp1 = (const float*)d_in[20]; const float* bp1 = (const float*)d_in[21];

    const int nT1 = BATCH * N1;   // 8192
    const int nT0 = BATCH * N0;   // 32768

    // workspace carve-out
    uintptr_t cur = (uintptr_t)d_ws;
    float*     p0      = (float*)    bump(cur, BATCH * 128 * sizeof(float));
    float*     p1      = (float*)    bump(cur, BATCH * 64  * sizeof(float));
    int*       idx1    = (int*)      bump(cur, (size_t)nT1 * 3 * sizeof(int));
    float*     w1      = (float*)    bump(cur, (size_t)nT1 * 4 * sizeof(float));
    int*       idx0    = (int*)      bump(cur, (size_t)nT0 * 3 * sizeof(int));
    float*     w0      = (float*)    bump(cur, (size_t)nT0 * 4 * sizeof(float));
    _Float16*  concat0 = (_Float16*) bump(cur, (size_t)nT1 * (C2 + C1) * sizeof(_Float16));
    _Float16*  h0      = (_Float16*) bump(cur, (size_t)nT1 * 256 * sizeof(_Float16));
    float*     x1      = (float*)    bump(cur, (size_t)nT1 * 128 * sizeof(float));
    _Float16*  concat1 = (_Float16*) bump(cur, (size_t)nT0 * (128 + C0) * sizeof(_Float16));
    _Float16*  h1      = (_Float16*) bump(cur, (size_t)nT0 * 128 * sizeof(_Float16));
    _Float16*  pW0a    = (_Float16*) bump(cur, (size_t)(C2 + C1) * 256 * sizeof(_Float16));
    _Float16*  pW0b    = (_Float16*) bump(cur, 256 * 128 * sizeof(_Float16));
    _Float16*  pW1a    = (_Float16*) bump(cur, 192 * 128 * sizeof(_Float16));
    _Float16*  pW1b    = (_Float16*) bump(cur, 128 * 64  * sizeof(_Float16));

    // weight packing (tiny; runs each call — deterministic)
    hipLaunchKernelGGL(pack_b_kernel, dim3((384 * 256 + 255) / 256), dim3(256), 0, stream, W0a, pW0a, 384, 256);
    hipLaunchKernelGGL(pack_b_kernel, dim3((256 * 128 + 255) / 256), dim3(256), 0, stream, W0b, pW0b, 256, 128);
    hipLaunchKernelGGL(pack_b_kernel, dim3((192 * 128 + 255) / 256), dim3(256), 0, stream, W1a, pW1a, 192, 128);
    hipLaunchKernelGGL(pack_b_kernel, dim3((128 * 64  + 255) / 256), dim3(256), 0, stream, W1b, pW1b, 128, 64);

    // per-sample parameter vectors
    hipLaunchKernelGGL(par_vec_kernel, dim3(3), dim3(256), 0, stream,
                       par, Wp0, bp0, Wp1, bp1, p0, p1);

    // ---- FP layer 0: coarse (N2) -> level-1 (N1) ----
    hipLaunchKernelGGL(knn3_kernel, dim3((nT1 + 255) / 256), dim3(256), 0, stream,
                       pos, pos_sk_l1, N2, N1, nT1, idx1, w1);
    {
        long tot = (long)nT1 * (C2 + C1);
        hipLaunchKernelGGL(interp_concat_kernel, dim3((unsigned)((tot + 255) / 256)), dim3(256), 0, stream,
                           x, C2, idx1, w1, x_skip_l1, C1, concat0, nT1);
    }
    // h0 = tanh(concat0 @ W0a + b0a)            M=8192 K=384 N=256
    hipLaunchKernelGGL(wmma_gemm_kernel, dim3((nT1 / 16) * (256 / 64) / 8), dim3(256), 0, stream,
                       concat0, pW0a, b0a, (const float*)nullptr, 0,
                       h0, (float*)nullptr, nT1, 256, 384, 1);
    // x1 = tanh(h0 @ W0b + b0b) * p0[sample]    M=8192 K=256 N=128
    hipLaunchKernelGGL(wmma_gemm_kernel, dim3((nT1 / 16) * (128 / 64) / 8), dim3(256), 0, stream,
                       h0, pW0b, b0b, p0, N1,
                       (_Float16*)nullptr, x1, nT1, 128, 256, 1);

    // ---- FP layer 1: level-1 (N1) -> level-0 (N0) ----
    hipLaunchKernelGGL(knn3_kernel, dim3((nT0 + 255) / 256), dim3(256), 0, stream,
                       pos_sk_l1, pos_sk_l0, N1, N0, nT0, idx0, w0);
    {
        long tot = (long)nT0 * (128 + C0);
        hipLaunchKernelGGL(interp_concat_kernel, dim3((unsigned)((tot + 255) / 256)), dim3(256), 0, stream,
                           x1, 128, idx0, w0, x_skip_l0, C0, concat1, nT0);
    }
    // h1 = tanh(concat1 @ W1a + b1a)            M=32768 K=192 N=128
    hipLaunchKernelGGL(wmma_gemm_kernel, dim3((nT0 / 16) * (128 / 64) / 8), dim3(256), 0, stream,
                       concat1, pW1a, b1a, (const float*)nullptr, 0,
                       h1, (float*)nullptr, nT0, 128, 192, 1);
    // out = (h1 @ W1b + b1b) * p1[sample]       M=32768 K=128 N=64 (plain_last)
    float* outX = (float*)d_out;
    hipLaunchKernelGGL(wmma_gemm_kernel, dim3((nT0 / 16) * (64 / 64) / 8), dim3(256), 0, stream,
                       h1, pW1b, b1b, p1, N0,
                       (_Float16*)nullptr, outX, nT0, 64, 128, 0);

    // ---- tuple tail: pos_skip_l0 (f32) and batch_skip_l0 (raw bits) ----
    float* outPos   = outX + (size_t)nT0 * 64;
    float* outBatch = outPos + (size_t)nT0 * 3;
    hipMemcpyAsync(outPos,   pos_sk_l0, (size_t)nT0 * 3 * sizeof(float), hipMemcpyDeviceToDevice, stream);
    hipMemcpyAsync(outBatch, batch_l0,  (size_t)nT0 * sizeof(int),       hipMemcpyDeviceToDevice, stream);
}
